// SpatioTemporalGCN_24816321036836
// MI455X (gfx1250) — matrix operand order repaired and checked by reading, not compiled, not run on hardware
//
#include <hip/hip_runtime.h>
#include <hip/hip_bf16.h>

typedef __attribute__((ext_vector_type(16))) _Float16 v16h;
typedef __attribute__((ext_vector_type(8)))  float    v8f;
typedef __attribute__((ext_vector_type(4)))  float    v4f;

#define N_NODES 8192
#define SEQ     256
#define HD      32
#define TCH     32
#define LOG2E   1.4426950408889634f
#define LN2     0.6931471805599453f

#if __has_builtin(__builtin_amdgcn_tanhf)
#define HAS_TANH 1
#else
#define HAS_TANH 0
#endif

__device__ __forceinline__ float fexp(float x) {
  return __builtin_amdgcn_exp2f(x * LOG2E);
}
__device__ __forceinline__ float ftanh(float x) {
#if HAS_TANH
  return __builtin_amdgcn_tanhf(x);
#else
  float t = __builtin_amdgcn_exp2f(2.0f * LOG2E * x);
  return (t - 1.0f) * __builtin_amdgcn_rcpf(t + 1.0f);
#endif
}
// sigmoid(z + b) with pre-halved bias bh = 0.5*b:  0.5*tanh(0.5*z + bh) + 0.5
__device__ __forceinline__ float fsigmoid_b(float z, float bh) {
#if HAS_TANH
  return fmaf(__builtin_amdgcn_tanhf(fmaf(z, 0.5f, bh)), 0.5f, 0.5f);
#else
  float x = fmaf(z, 2.0f, 4.0f * bh);
  return __builtin_amdgcn_rcpf(1.0f + __builtin_amdgcn_exp2f(-x * (0.5f * LOG2E)));
#endif
}

__device__ __forceinline__ v8f wmma_f16(v16h a, v16h b, v8f c) {
  // D = A(16x32 f16) x B(32x16 f16) + C(16x16 f32)
  return __builtin_amdgcn_wmma_f32_16x16x32_f16(
      false, a, false, b, (short)0, c, false, false);
}

// A-fragment (16x32 f16) from row-major [16][32] f16 LDS tile.
// Lane L: M=L&15; lanes 0-15: K=0..7,16..23; lanes 16-31: K=8..15,24..31.
__device__ __forceinline__ v16h load_afrag(const _Float16* buf, int lane) {
  const int row = lane & 15;
  const int kb  = (lane >> 4) << 3;
  union { v4f f[2]; v16h h; } u;
  const v4f* p = (const v4f*)(buf + row * 32 + kb);
  u.f[0] = p[0];
  u.f[1] = p[2];
  return u.h;
}

// B-fragment stored frag-linear in LDS: 32 bytes per lane per tile.
__device__ __forceinline__ v16h load_bfrag(const _Float16* tile, int lane) {
  union { v4f f[2]; v16h h; } u;
  const v4f* p = (const v4f*)(tile + lane * 16);
  u.f[0] = p[0]; u.f[1] = p[1];
  return u.h;
}
__device__ __forceinline__ void store_bfrag(_Float16* tile, int lane, v16h frag) {
  union { v16h h; v4f f[2]; } s; s.h = frag;
  v4f* p = (v4f*)(tile + lane * 16);
  p[0] = s.f[0]; p[1] = s.f[1];
}

__device__ __forceinline__ uint32_t pack2h(float a, float b) {
  union { _Float16 h[2]; uint32_t u; } p;
  p.h[0] = (_Float16)a; p.h[1] = (_Float16)b;
  return p.u;
}

// ---- CDNA5 async global->LDS copy (ASYNCcnt-tracked) ----
__device__ __forceinline__ void async_ld_f32(const float* gp, const void* ldsp) {
  asm volatile("global_load_async_to_lds_b32 %0, %1, off"
               :: "v"((uint32_t)(uintptr_t)ldsp),
                  "v"((unsigned long long)(uintptr_t)gp)
               : "memory");
}
__device__ __forceinline__ void wait_async0() {
  asm volatile("s_wait_asynccnt 0" ::: "memory");
}

// -------------------- Fused LSTM1 + LayerNorm + LSTM2 --------------------
// One wave = 16 nodes.  Block = 8 waves = 128 nodes.  Grid = 64 blocks.
// Hidden-unit remap: lane u owns units (2u, 2u+1).
// All 24 weight B-fragments live in a block-shared LDS pool (one base reg,
// immediate offsets) -> VGPR pressure < 256, no s_set_vgpr_msb in hot loop.
__global__ __launch_bounds__(256)
void lstm_fused(const float* __restrict__ x,
                const float* __restrict__ Wih1, const float* __restrict__ Whh1,
                const float* __restrict__ bih1, const float* __restrict__ bhh1,
                const float* __restrict__ lng,  const float* __restrict__ lnb,
                const float* __restrict__ Wih2, const float* __restrict__ Whh2,
                const float* __restrict__ bih2, const float* __restrict__ bhh2,
                float* __restrict__ h2out) {
  __shared__ __align__(64) float    xs[8][2][TCH][16];   // double-buffered x
  __shared__ __align__(64) _Float16 h1s[8][16][32];
  __shared__ __align__(64) _Float16 ys [8][16][32];
  __shared__ __align__(64) _Float16 h2s[8][16][32];
  __shared__ __align__(64) _Float16 wfrag[24][512];      // 0-7 Whh1, 8-15 Wih2, 16-23 Whh2

  const int tid  = threadIdx.x;
  const int w    = tid >> 5;
  const int lane = tid & 31;
  const int u    = lane & 15;
  const int hi   = lane >> 4;
  const int nodeBase = blockIdx.x * 128 + w * 16;
  const int kb = hi * 8;

  // async prefetch of x chunk 0 (lane = t within chunk, coalesced)
#pragma unroll
  for (int m = 0; m < 16; ++m)
    async_ld_f32(&x[(nodeBase + m) * SEQ + lane], &xs[w][0][lane][m]);

  // ---- per-lane constants ----
  // Tile j = gate*2 + p covers gate column  gate*32 + 2*u + p.
  // bias:  gates i,f,o pre-halved (sigmoid fold); gate g full.
  float wih1v[8], bias1[8], bias2[8];
#pragma unroll
  for (int j = 0; j < 8; ++j) {
    const int gcol = (j >> 1) * 32 + 2 * u + (j & 1);
    const float sc = ((j >> 1) == 2) ? 1.0f : 0.5f;
    wih1v[j] = Wih1[gcol];
    bias1[j] = sc * (bih1[gcol] + bhh1[gcol]);
    bias2[j] = sc * (bih2[gcol] + bhh2[gcol]);
  }
  // waves 0..2 publish the three weight-fragment groups to shared LDS
  if (w < 3) {
    const float* Wsrc = (w == 0) ? Whh1 : (w == 1) ? Wih2 : Whh2;
#pragma unroll
    for (int j = 0; j < 8; ++j) {
      const int gcol = (j >> 1) * 32 + 2 * u + (j & 1);
      const float* wp = Wsrc + gcol * HD + kb;
      v16h f;
#pragma unroll
      for (int e = 0; e < 16; ++e) f[e] = (_Float16)wp[e + (e & 8)];
      store_bfrag(wfrag[w * 8 + j], lane, f);
    }
  }
  const float lng0 = lng[2 * u], lng1 = lng[2 * u + 1];
  const float lnb0 = lnb[2 * u], lnb1 = lnb[2 * u + 1];

  v16h onesF;
#pragma unroll
  for (int e = 0; e < 16; ++e) onesF[e] = (_Float16)1.0f;

  __syncthreads();

  float c1[2][8], c2[2][8], h2n[2][8];
#pragma unroll
  for (int q = 0; q < 2; ++q)
#pragma unroll
    for (int r = 0; r < 8; ++r) { c1[q][r] = 0.f; c2[q][r] = 0.f; h2n[q][r] = 0.f; }
  v16h a1 = {};   // carried h1 A-fragment
  v16h a2 = {};   // carried h2 A-fragment

  for (int t = 0; t < SEQ; ++t) {
    const int tm  = t & (TCH - 1);
    const int buf = (t >> 5) & 1;
    if (tm == 0) {
      wait_async0();                       // current chunk resident
      if (t + TCH < SEQ) {
#pragma unroll
        for (int m = 0; m < 16; ++m)
          async_ld_f32(&x[(nodeBase + m) * SEQ + (t + TCH) + lane],
                       &xs[w][buf ^ 1][lane][m]);
      }
    }

    // x_t per C-row: VGPR r <-> node m = r + hi*8
    v4f xv0 = *(const v4f*)&xs[w][buf][tm][hi * 8];
    v4f xv1 = *(const v4f*)&xs[w][buf][tm][hi * 8 + 4];
    float xv[8] = {xv0[0], xv0[1], xv0[2], xv0[3], xv1[0], xv1[1], xv1[2], xv1[3]};

    // ---------------- LSTM layer 1 ----------------
    float h1n[2][8];
#pragma unroll
    for (int p = 0; p < 2; ++p) {
      v8f ci, cg;
#pragma unroll
      for (int r = 0; r < 8; ++r) {
        ci[r] = xv[r] * wih1v[0 + p];
        cg[r] = xv[r] * wih1v[4 + p];
      }
      v8f zi = wmma_f16(a1, load_bfrag(wfrag[0 + p], lane), ci);
      v8f zg = wmma_f16(a1, load_bfrag(wfrag[4 + p], lane), cg);
      float ig[8];
#pragma unroll
      for (int r = 0; r < 8; ++r)
        ig[r] = fsigmoid_b(zi[r], bias1[0 + p]) * ftanh(zg[r] + bias1[4 + p]);
      v8f cf, co;
#pragma unroll
      for (int r = 0; r < 8; ++r) {
        cf[r] = xv[r] * wih1v[2 + p];
        co[r] = xv[r] * wih1v[6 + p];
      }
      v8f zf = wmma_f16(a1, load_bfrag(wfrag[2 + p], lane), cf);
      v8f zo = wmma_f16(a1, load_bfrag(wfrag[6 + p], lane), co);
#pragma unroll
      for (int r = 0; r < 8; ++r) {
        float c = fmaf(fsigmoid_b(zf[r], bias1[2 + p]), c1[p][r], ig[r]);
        c1[p][r] = c;
        h1n[p][r] = fsigmoid_b(zo[r], bias1[6 + p]) * ftanh(c);
      }
    }
    // packed store of new h1; reload as next A-fragment
#pragma unroll
    for (int r = 0; r < 8; ++r)
      ((uint32_t*)&h1s[w][r + hi * 8][0])[u] = pack2h(h1n[0][r], h1n[1][r]);
    a1 = load_afrag(&h1s[w][0][0], lane);

    // ---------------- LayerNorm via WMMA row-sums ----------------
    // S = h1 x ones, Q = h1^2 x ones: every column of D equals the row sum,
    // landing S[m=r+hi*8] exactly in each lane's C-layout slot r.
    {
      v16h a1sq = a1 * a1;                 // v_pk_mul_f16
      v8f zc = {};
      v8f S = wmma_f16(a1, onesF, zc);
      v8f Q = wmma_f16(a1sq, onesF, zc);
#pragma unroll
      for (int r = 0; r < 8; ++r) {
        float mu  = S[r] * (1.0f / 32.0f);
        float var = fmaf(Q[r], 1.0f / 32.0f, -mu * mu);
        float inv = __builtin_amdgcn_rsqf(var + 1e-5f);
        float y0 = (h1n[0][r] - mu) * inv * lng0 + lnb0;
        float y1 = (h1n[1][r] - mu) * inv * lng1 + lnb1;
        ((uint32_t*)&ys[w][r + hi * 8][0])[u] = pack2h(y0, y1);
      }
    }

    // ---------------- LSTM layer 2 ----------------
    v16h ay = load_afrag(&ys[w][0][0], lane);
#pragma unroll
    for (int p = 0; p < 2; ++p) {
      v8f zc = {};
      v8f zi = wmma_f16(a2, load_bfrag(wfrag[16 + 0 + p], lane),
                        wmma_f16(ay, load_bfrag(wfrag[8 + 0 + p], lane), zc));
      v8f zg = wmma_f16(a2, load_bfrag(wfrag[16 + 4 + p], lane),
                        wmma_f16(ay, load_bfrag(wfrag[8 + 4 + p], lane), zc));
      float ig[8];
#pragma unroll
      for (int r = 0; r < 8; ++r)
        ig[r] = fsigmoid_b(zi[r], bias2[0 + p]) * ftanh(zg[r] + bias2[4 + p]);
      v8f zf = wmma_f16(a2, load_bfrag(wfrag[16 + 2 + p], lane),
                        wmma_f16(ay, load_bfrag(wfrag[8 + 2 + p], lane), zc));
      v8f zo = wmma_f16(a2, load_bfrag(wfrag[16 + 6 + p], lane),
                        wmma_f16(ay, load_bfrag(wfrag[8 + 6 + p], lane), zc));
#pragma unroll
      for (int r = 0; r < 8; ++r) {
        float c = fmaf(fsigmoid_b(zf[r], bias2[2 + p]), c2[p][r], ig[r]);
        c2[p][r] = c;
        h2n[p][r] = fsigmoid_b(zo[r], bias2[6 + p]) * ftanh(c);
      }
    }
#pragma unroll
    for (int r = 0; r < 8; ++r)
      ((uint32_t*)&h2s[w][r + hi * 8][0])[u] = pack2h(h2n[0][r], h2n[1][r]);
    a2 = load_afrag(&h2s[w][0][0], lane);
  }

  // final h2 (fp32, packed b64) -> gcn_in
#pragma unroll
  for (int r = 0; r < 8; ++r) {
    const int node = nodeBase + r + hi * 8;
    float2 o; o.x = h2n[0][r]; o.y = h2n[1][r];
    *(float2*)&h2out[node * HD + 2 * u] = o;
  }
}

// ------------------------------ GCN tail ------------------------------
__global__ void k_fill(float* p, float v, int n) {
  int i = blockIdx.x * blockDim.x + threadIdx.x;
  if (i < n) p[i] = v;
}

__global__ void k_deg(const int* __restrict__ ei, const float* __restrict__ ew,
                      float* deg, int E) {
  int e = blockIdx.x * blockDim.x + threadIdx.x;
  if (e < E) atomicAdd(&deg[ei[E + e]], ew[e]);   // dst row
}

__global__ void k_dinv(const float* __restrict__ deg, float* dinv, int n) {
  int i = blockIdx.x * blockDim.x + threadIdx.x;
  if (i < n) dinv[i] = __builtin_amdgcn_rsqf(deg[i]);   // deg >= 1
}

__global__ __launch_bounds__(256)
void k_gemm32(const float* __restrict__ X, const float* __restrict__ W,
              float* __restrict__ Y) {
  // Y[n][j] = sum_k X[n][k] * W[k][j]
  __shared__ float Ws[HD * HD];
#pragma unroll
  for (int i = 0; i < 4; ++i) Ws[threadIdx.x + 256 * i] = W[threadIdx.x + 256 * i];
  __syncthreads();
  int idx = blockIdx.x * blockDim.x + threadIdx.x;
  int n = idx >> 5, j = idx & 31;
  const float* xr = X + n * HD;
  float acc = 0.f;
#pragma unroll
  for (int k = 0; k < HD; ++k) acc = fmaf(xr[k], Ws[k * HD + j], acc);
  Y[idx] = acc;
}

__global__ void k_selfbias(const float* __restrict__ h, const float* __restrict__ dinv,
                           const float* __restrict__ bias, float* __restrict__ agg) {
  int idx = blockIdx.x * blockDim.x + threadIdx.x;
  int n = idx >> 5, kk = idx & 31;
  float dv = dinv[n];
  agg[idx] = h[idx] * dv * dv + bias[kk];
}

__global__ void k_scatter(const int* __restrict__ ei, const float* __restrict__ ew,
                          const float* __restrict__ dinv, const float* __restrict__ h,
                          float* __restrict__ agg, int E) {
  int idx = blockIdx.x * blockDim.x + threadIdx.x;   // E * 32 threads
  int e = idx >> 5, kk = idx & 31;
  if (e < E) {
    int s = ei[e], d = ei[E + e];
    float norm = dinv[s] * ew[e] * dinv[d];
    atomicAdd(&agg[d * HD + kk], norm * h[s * HD + kk]);
  }
}

__global__ void k_elu(float* p, int n) {
  int i = blockIdx.x * blockDim.x + threadIdx.x;
  if (i < n) {
    float v = p[i];
    p[i] = v > 0.f ? v : (fexp(v) - 1.0f);
  }
}

__global__ void k_final(const float* __restrict__ gcn_in, const float* __restrict__ g2,
                        const float* __restrict__ Wfc, const float* __restrict__ bfc,
                        float* __restrict__ out) {
  int n = blockIdx.x * blockDim.x + threadIdx.x;
  if (n < N_NODES) {
    float mg = 0.f;
#pragma unroll
    for (int k = 0; k < HD; ++k) mg += g2[n * HD + k];
    mg *= (1.0f / 32.0f);
    float o0 = bfc[0] + Wfc[HD] * mg;          // row 0: Wfc[0..32]
    float o1 = bfc[1] + Wfc[33 + HD] * mg;     // row 1: Wfc[33..65]
#pragma unroll
    for (int k = 0; k < HD; ++k) {
      float v = gcn_in[n * HD + k];
      o0 = fmaf(Wfc[k], v, o0);
      o1 = fmaf(Wfc[33 + k], v, o1);
    }
    float m = fmaxf(o0, o1);
    float l = m + LN2 * __builtin_amdgcn_logf(fexp(o0 - m) + fexp(o1 - m));
    out[n * 2 + 0] = o0 - l;
    out[n * 2 + 1] = o1 - l;
  }
}

// ------------------------------ launcher ------------------------------
extern "C" void kernel_launch(void* const* d_in, const int* in_sizes, int n_in,
                              void* d_out, int out_size, void* d_ws, size_t ws_size,
                              hipStream_t stream) {
  const float* x    = (const float*)d_in[0];
  const float* ew   = (const float*)d_in[1];
  const float* Wih1 = (const float*)d_in[2];
  const float* Whh1 = (const float*)d_in[3];
  const float* bih1 = (const float*)d_in[4];
  const float* bhh1 = (const float*)d_in[5];
  const float* lng  = (const float*)d_in[6];
  const float* lnb  = (const float*)d_in[7];
  const float* Wih2 = (const float*)d_in[8];
  const float* Whh2 = (const float*)d_in[9];
  const float* bih2 = (const float*)d_in[10];
  const float* bhh2 = (const float*)d_in[11];
  const float* Wg1  = (const float*)d_in[12];
  const float* bg1  = (const float*)d_in[13];
  const float* Wg2  = (const float*)d_in[14];
  const float* bg2  = (const float*)d_in[15];
  const float* Wfc  = (const float*)d_in[16];
  const float* bfc  = (const float*)d_in[17];
  const int*   ei   = (const int*)d_in[18];
  const int E = in_sizes[1];
  float* out = (float*)d_out;

  float* ws     = (float*)d_ws;
  float* gcn_in = ws;                       // [N,32]
  float* B1     = gcn_in + N_NODES * HD;    // temp h = X@W
  float* B2     = B1 + N_NODES * HD;        // g (conv1 out)
  float* B3     = B2 + N_NODES * HD;        // g2 (conv2 out)
  float* deg    = B3 + N_NODES * HD;        // [N]
  float* dinv   = deg + N_NODES;            // [N]

  // fused LSTM1 + LN + LSTM2  -> gcn_in
  lstm_fused<<<N_NODES / 128, 256, 0, stream>>>(
      x, Wih1, Whh1, bih1, bhh1, lng, lnb, Wih2, Whh2, bih2, bhh2, gcn_in);

  // degree normalization
  k_fill<<<(N_NODES + 255) / 256, 256, 0, stream>>>(deg, 1.0f, N_NODES);
  k_deg<<<(E + 255) / 256, 256, 0, stream>>>(ei, ew, deg, E);
  k_dinv<<<(N_NODES + 255) / 256, 256, 0, stream>>>(deg, dinv, N_NODES);

  const int NW = N_NODES * HD;
  // GCN conv 1
  k_gemm32<<<NW / 256, 256, 0, stream>>>(gcn_in, Wg1, B1);
  k_selfbias<<<NW / 256, 256, 0, stream>>>(B1, dinv, bg1, B2);
  k_scatter<<<(E * HD) / 256, 256, 0, stream>>>(ei, ew, dinv, B1, B2, E);
  k_elu<<<NW / 256, 256, 0, stream>>>(B2, NW);
  // GCN conv 2
  k_gemm32<<<NW / 256, 256, 0, stream>>>(B2, Wg2, B1);
  k_selfbias<<<NW / 256, 256, 0, stream>>>(B1, dinv, bg2, B3);
  k_scatter<<<(E * HD) / 256, 256, 0, stream>>>(ei, ew, dinv, B1, B3, E);
  k_elu<<<NW / 256, 256, 0, stream>>>(B3, NW);
  // FC + log_softmax
  k_final<<<(N_NODES + 255) / 256, 256, 0, stream>>>(gcn_in, B3, Wfc, bfc, out);
}
